// RecurrentPolicy_9423158248001
// MI455X (gfx1250) — matrix operand order repaired and checked by reading, not compiled, not run on hardware
//
#include <hip/hip_runtime.h>
#include <hip/hip_bf16.h>
#include <math.h>

// ---------------- problem dims ----------------
#define BB   256   // batch
#define TT   512   // time
#define NOBS 128
#define HH   128   // hidden (MEM/2)
#define GG   512   // 4*H gates
#define INX  384   // OBS + MEM
#define H1   256
#define H2   256
#define ACT  32

typedef __attribute__((ext_vector_type(16))) _Float16 v16h;
typedef __attribute__((ext_vector_type(8)))  _Float16 v8h;
typedef __attribute__((ext_vector_type(4)))  _Float16 v4h;
typedef __attribute__((ext_vector_type(8)))  float    v8f;

union V16U { v16h v; v8h h[2]; };

__device__ __forceinline__ v8f splat8(float x) {
    v8f v = {x, x, x, x, x, x, x, x};
    return v;
}

__device__ __forceinline__ float sigmoidf(float x) {
    return 1.0f / (1.0f + __expf(-x));
}

// Issue one 16B async global->LDS copy (CDNA5 GLOBAL_LOAD_ASYNC_TO_LDS_B128,
// tracked by ASYNCcnt). ldsaddr = wave-relative LDS byte offset (low 32 bits
// of the generic pointer, per the LDS aperture mapping).
__device__ __forceinline__ void async_copy_b128(const void* gptr, void* lptr) {
    unsigned           l = (unsigned)(uintptr_t)lptr;
    unsigned long long g = (unsigned long long)(uintptr_t)gptr;
    asm volatile("global_load_async_to_lds_b128 %0, %1, off"
                 :: "v"(l), "v"(g) : "memory");
}

__device__ __forceinline__ void wait_async0() {
    asm volatile("s_wait_asynccnt 0" ::: "memory");
}

// vectorized f32->f16 LDS-to-LDS conversion: n4 float4 chunks, 256 threads
__device__ __forceinline__ void cvt_tile_lds(const float* src, _Float16* dst,
                                             int n4, int tid) {
    for (int c = tid; c < n4; c += 256) {
        const float4 v = ((const float4*)src)[c];
        v4h h = {(_Float16)v.x, (_Float16)v.y, (_Float16)v.z, (_Float16)v.w};
        ((v4h*)dst)[c] = h;
    }
}

// A fragment (16x32 f16, M x K) from row-major tile base with leading dim ld.
// ISA layout: lanes 0-15 -> elems 0..7 = K(k0+0..7), 8..15 = K(k0+16..23)
//             lanes 16-31 -> elems 0..7 = K(k0+8..15), 8..15 = K(k0+24..31)
__device__ __forceinline__ v16h loadA(const _Float16* base, int ld, int k0, int lane) {
    const int m  = lane & 15;
    const int hi = lane >> 4;
    const _Float16* p = base + m * ld + k0 + hi * 8;
    V16U u;
    u.h[0] = *(const v8h*)(p);
    u.h[1] = *(const v8h*)(p + 16);
    return u.v;
}

// B fragment (32x16 f16, K x N) taken from row-major weight W[N][K] (ld = K
// stride), i.e. B[k][n] = W[n0+n][k0+k]. ISA layout: lane n = lane&15,
// elems 0..7 = K(k0 + hi*8 + 0..7), 8..15 = K(k0 + 16 + hi*8 + 0..7).
__device__ __forceinline__ v16h loadB(const _Float16* W, int ld, int n0, int k0, int lane) {
    const int n  = lane & 15;
    const int hi = lane >> 4;
    const _Float16* p = W + (size_t)(n0 + n) * ld + k0 + hi * 8;
    V16U u;
    u.h[0] = *(const v8h*)(p);
    u.h[1] = *(const v8h*)(p + 16);
    return u.v;
}

__device__ __forceinline__ v8f wmma_f16(v16h a, v16h b, v8f c) {
    return __builtin_amdgcn_wmma_f32_16x16x32_f16(
        /*neg_a=*/false, a, /*neg_b=*/false, b,
        /*c_mod=*/(short)0, c, /*reuse_a=*/false, /*reuse_b=*/false);
}

// ---------------- weight conversion ----------------
__global__ void cvt_f32_f16_kernel(const float* __restrict__ src,
                                   _Float16* __restrict__ dst, int n) {
    int i = blockIdx.x * blockDim.x + threadIdx.x;
    if (i < n) dst[i] = (_Float16)src[i];
}

__global__ void bias_sum_kernel(const float* __restrict__ a,
                                const float* __restrict__ b,
                                float* __restrict__ o, int n) {
    int i = blockIdx.x * blockDim.x + threadIdx.x;
    if (i < n) o[i] = a[i] + b[i];
}

// ---------------- LSTM kernel ----------------
// grid = B/16 workgroups, 256 threads (8 waves). Wave w owns h-columns
// [16w, 16w+16). Each wave computes the 4 gate tiles (i,f,g,o) for its
// column slice: 4 tiles x (4 x-WMMAs + 4 h-WMMAs) = 32 WMMA/step.
// x_t tiles are double-buffered into LDS via GLOBAL_LOAD_ASYNC_TO_LDS_B128
// so global latency is off the serial critical path; weights live in the
// extended (1024) VGPR file after the first iteration.
__global__ __launch_bounds__(256) void lstm_kernel(
    const float* __restrict__ obs,        // (B,T,OBS)
    const float* __restrict__ init_mem,   // (B,2H)
    const _Float16* __restrict__ Wih,     // (4H,OBS) f16 row-major
    const _Float16* __restrict__ Whh,     // (4H,H)   f16 row-major
    const float* __restrict__ bg,         // (4H) = b_ih + b_hh
    float* __restrict__ writes)           // (B,T,2H)
{
    __shared__ __align__(16) float    xf32[2][16 * NOBS];  // async staging, 2x8KB
    __shared__ __align__(16) _Float16 xs[16 * NOBS];       // x_t tile f16
    __shared__ __align__(16) _Float16 hsl[16 * HH];        // h tile f16
    __shared__ float bgs[GG];

    const int tid  = threadIdx.x;
    const int lane = tid & 31;
    const int w    = tid >> 5;      // wave id 0..7
    const int hi   = lane >> 4;
    const int nn   = lane & 15;
    const int batch0 = blockIdx.x * 16;
    const int colh = w * 16 + nn;   // this lane's h column

    // issue async loads for t=0 tile: 512 x 16B chunks, 2 per thread
    #pragma unroll
    for (int i = 0; i < 2; ++i) {
        const int c = tid + i * 256;
        const int m = c >> 5;       // row in tile
        const int q = c & 31;       // 16B chunk within row
        async_copy_b128(obs + ((size_t)(batch0 + m) * TT + 0) * NOBS + q * 4,
                        &xf32[0][m * NOBS + q * 4]);
    }

    for (int i = tid; i < GG; i += 256) bgs[i] = bg[i];

    // init h, c in WMMA C/D fragment layout: elem r -> (m = r+hi*8, colh)
    v8f hfrag, cfrag;
    #pragma unroll
    for (int r = 0; r < 8; ++r) {
        const int b = batch0 + r + hi * 8;
        hfrag[r] = init_mem[(size_t)b * 256 + colh];
        cfrag[r] = init_mem[(size_t)b * 256 + HH + colh];
    }

    for (int t = 0; t < TT; ++t) {
        wait_async0();          // my async writes for step t are in LDS
        __syncthreads();        // everyone's are visible; prev WMMA reads done

        // convert x_t tile f32 -> f16 (LDS-local, vectorized)
        cvt_tile_lds(xf32[t & 1], xs, (16 * NOBS) / 4, tid);
        // publish h for this step's WMMA
        #pragma unroll
        for (int r = 0; r < 8; ++r)
            hsl[(r + hi * 8) * HH + colh] = (_Float16)hfrag[r];
        // kick off async loads for t+1 into the other buffer
        if (t + 1 < TT) {
            float* nxt = xf32[(t + 1) & 1];
            #pragma unroll
            for (int i = 0; i < 2; ++i) {
                const int c = tid + i * 256;
                const int m = c >> 5;
                const int q = c & 31;
                async_copy_b128(obs + ((size_t)(batch0 + m) * TT + (t + 1)) * NOBS + q * 4,
                                nxt + m * NOBS + q * 4);
            }
        }
        __syncthreads();

        v8f acc[4];
        #pragma unroll
        for (int g = 0; g < 4; ++g) {
            const int n0 = g * HH + w * 16;
            v8f a = splat8(bgs[n0 + nn]);
            #pragma unroll
            for (int k0 = 0; k0 < 128; k0 += 32) {
                a = wmma_f16(loadA(xs,  NOBS, k0, lane), loadB(Wih, NOBS, n0, k0, lane), a);
                a = wmma_f16(loadA(hsl, HH,   k0, lane), loadB(Whh, HH,   n0, k0, lane), a);
            }
            acc[g] = a;
        }

        // elementwise LSTM cell + write h,c (this step's "writes")
        #pragma unroll
        for (int r = 0; r < 8; ++r) {
            const float ig = sigmoidf(acc[0][r]);
            const float fg = sigmoidf(acc[1][r]);
            const float gg = tanhf(acc[2][r]);
            const float og = sigmoidf(acc[3][r]);
            const float cn = fg * cfrag[r] + ig * gg;
            const float hn = og * tanhf(cn);
            cfrag[r] = cn;
            hfrag[r] = hn;
            const size_t orow = ((size_t)(batch0 + r + hi * 8) * TT + t) * 256;
            writes[orow + colh]      = hn;
            writes[orow + HH + colh] = cn;
        }
        // top-of-loop barrier orders these stores/reads vs next step
    }
}

// ---------------- fused MLP kernel ----------------
// One WG = 32 rows (two 16-row M-tiles) of (B*T). Each weight B fragment is
// loaded once per wave and reused for both M-tiles -> halves L2 weight
// traffic vs 16-row tiles. x async-staged into LDS (f32), converted to f16;
// three layers fused, activations never leave LDS. LDS = 104 KB (3 WG/WGP).
__global__ __launch_bounds__(256) void mlp_kernel(
    const float* __restrict__ obs,       // (B*T,128)
    const float* __restrict__ init_mem,  // (B,256)
    const float* __restrict__ writes,    // (B*T,256)
    const _Float16* __restrict__ W1, const float* __restrict__ b1,
    const _Float16* __restrict__ W2, const float* __restrict__ b2,
    const _Float16* __restrict__ Wa, const float* __restrict__ ba,
    float* __restrict__ actions)         // (B*T,32)
{
    __shared__ __align__(16) float    xf[32 * INX];   // 48 KB async staging
    __shared__ __align__(16) _Float16 xsl[32 * INX];  // 24 KB
    __shared__ __align__(16) _Float16 h1s[32 * H1];   // 16 KB
    __shared__ __align__(16) _Float16 h2s[32 * H2];   // 16 KB

    const int tid  = threadIdx.x;
    const int lane = tid & 31;
    const int w    = tid >> 5;
    const int hi   = lane >> 4;
    const int nn   = lane & 15;
    const size_t row0 = (size_t)blockIdx.x * 32;

    // async-stage x tile: 32 rows x 384 f32 = 3072 x 16B chunks, 12/thread.
    // Per-lane source select handles the memory shift (t==0 -> initial_memory).
    #pragma unroll
    for (int i = 0; i < 12; ++i) {
        const int c = tid + i * 256;
        const int m = c / 96;            // 96 chunks per row
        const int q = c - m * 96;
        const int k = q * 4;             // float index in row
        const size_t rr = row0 + m;
        const float* g;
        if (k < NOBS) {
            g = obs + rr * NOBS + k;
        } else {
            const int j = k - NOBS;
            const int t = (int)(rr & (TT - 1));          // T = 512 pow2
            g = (t == 0) ? init_mem + (rr >> 9) * 256 + j
                         : writes + (rr - 1) * 256 + j;
        }
        async_copy_b128(g, &xf[m * INX + k]);
    }
    wait_async0();
    __syncthreads();
    cvt_tile_lds(xf, xsl, (32 * INX) / 4, tid);
    __syncthreads();

    // layer 1: 256 outputs, wave w -> n0 = w*16 and 128+w*16, K=384.
    // B fragment loaded once, used for both M-tiles.
    #pragma unroll
    for (int half = 0; half < 2; ++half) {
        const int n0 = half * 128 + w * 16;
        const float bb = b1[n0 + nn];
        v8f a0 = splat8(bb), a1 = splat8(bb);
        #pragma unroll
        for (int k0 = 0; k0 < INX; k0 += 32) {
            const v16h b = loadB(W1, INX, n0, k0, lane);
            a0 = wmma_f16(loadA(xsl,            INX, k0, lane), b, a0);
            a1 = wmma_f16(loadA(xsl + 16 * INX, INX, k0, lane), b, a1);
        }
        #pragma unroll
        for (int r = 0; r < 8; ++r) {
            h1s[(r + hi * 8) * H1 + n0 + nn]        = (_Float16)fmaxf(a0[r], 0.0f);
            h1s[(16 + r + hi * 8) * H1 + n0 + nn]   = (_Float16)fmaxf(a1[r], 0.0f);
        }
    }
    __syncthreads();

    // layer 2: 256 outputs, K=256
    #pragma unroll
    for (int half = 0; half < 2; ++half) {
        const int n0 = half * 128 + w * 16;
        const float bb = b2[n0 + nn];
        v8f a0 = splat8(bb), a1 = splat8(bb);
        #pragma unroll
        for (int k0 = 0; k0 < H1; k0 += 32) {
            const v16h b = loadB(W2, H1, n0, k0, lane);
            a0 = wmma_f16(loadA(h1s,           H1, k0, lane), b, a0);
            a1 = wmma_f16(loadA(h1s + 16 * H1, H1, k0, lane), b, a1);
        }
        #pragma unroll
        for (int r = 0; r < 8; ++r) {
            h2s[(r + hi * 8) * H2 + n0 + nn]      = (_Float16)fmaxf(a0[r], 0.0f);
            h2s[(16 + r + hi * 8) * H2 + n0 + nn] = (_Float16)fmaxf(a1[r], 0.0f);
        }
    }
    __syncthreads();

    // layer 3: 32 outputs x 2 M-tiles -> waves 0..3 (wave-uniform branch)
    if (w < 4) {
        const int mt = w >> 1;
        const int n0 = (w & 1) * 16;
        v8f a = splat8(ba[n0 + nn]);
        #pragma unroll
        for (int k0 = 0; k0 < H2; k0 += 32)
            a = wmma_f16(loadA(h2s + mt * 16 * H2, H2, k0, lane),
                         loadB(Wa, H2, n0, k0, lane), a);
        #pragma unroll
        for (int r = 0; r < 8; ++r)
            actions[(row0 + mt * 16 + r + hi * 8) * ACT + n0 + nn] = tanhf(a[r]);
    }
}

// ---------------- launcher ----------------
extern "C" void kernel_launch(void* const* d_in, const int* in_sizes, int n_in,
                              void* d_out, int out_size, void* d_ws, size_t ws_size,
                              hipStream_t stream) {
    const float* obs      = (const float*)d_in[0];
    const float* init_mem = (const float*)d_in[1];
    const float* W_ih     = (const float*)d_in[2];
    const float* W_hh     = (const float*)d_in[3];
    const float* b_ih     = (const float*)d_in[4];
    const float* b_hh     = (const float*)d_in[5];
    const float* W1       = (const float*)d_in[6];
    const float* b1       = (const float*)d_in[7];
    const float* W2       = (const float*)d_in[8];
    const float* b2       = (const float*)d_in[9];
    const float* Wa       = (const float*)d_in[10];
    const float* ba       = (const float*)d_in[11];

    // workspace layout (f16 element offsets)
    _Float16* ws16  = (_Float16*)d_ws;
    _Float16* Wih16 = ws16 + 0;                // 512*128
    _Float16* Whh16 = ws16 + 65536;            // 512*128
    _Float16* W1_16 = ws16 + 131072;           // 256*384
    _Float16* W2_16 = ws16 + 229376;           // 256*256
    _Float16* Wa_16 = ws16 + 294912;           // 32*256
    float*    bg    = (float*)((char*)d_ws + 606208);  // 512 f32

    float* actions = (float*)d_out;                          // (B*T,32)
    float* writes  = (float*)d_out + (size_t)BB * TT * ACT;  // (B*T,256)

    const int cvt_b = 256;
    cvt_f32_f16_kernel<<<(65536 + cvt_b - 1) / cvt_b, cvt_b, 0, stream>>>(W_ih, Wih16, 65536);
    cvt_f32_f16_kernel<<<(65536 + cvt_b - 1) / cvt_b, cvt_b, 0, stream>>>(W_hh, Whh16, 65536);
    cvt_f32_f16_kernel<<<(98304 + cvt_b - 1) / cvt_b, cvt_b, 0, stream>>>(W1, W1_16, 98304);
    cvt_f32_f16_kernel<<<(65536 + cvt_b - 1) / cvt_b, cvt_b, 0, stream>>>(W2, W2_16, 65536);
    cvt_f32_f16_kernel<<<(8192  + cvt_b - 1) / cvt_b, cvt_b, 0, stream>>>(Wa, Wa_16, 8192);
    bias_sum_kernel<<<2, 256, 0, stream>>>(b_ih, b_hh, bg, GG);

    lstm_kernel<<<BB / 16, 256, 0, stream>>>(obs, init_mem, Wih16, Whh16, bg, writes);

    mlp_kernel<<<(BB * TT) / 32, 256, 0, stream>>>(obs, init_mem, writes,
                                                   W1_16, b1, W2_16, b2,
                                                   Wa_16, ba, actions);
}
